// EGNN_dynamics_QM9_65317862638365
// MI455X (gfx1250) — compile-verified, matrix operand order checked by reading, not run on hardware
//
#include <hip/hip_runtime.h>
#include <hip/hip_bf16.h>

typedef __attribute__((ext_vector_type(16))) _Float16 v16h;
typedef __attribute__((ext_vector_type(8)))  _Float16 v8h;
typedef __attribute__((ext_vector_type(8)))  float    v8f;

#define BSZ 64
#define NN 44
#define NPAD 48
#define HD 256
#define NLAYERS 4
#define INV_SUB 2
#define GG (NLAYERS*INV_SUB)
#define REP_NF 512
#define INV_NORM 0.01f

// ---------------------------------------------------------------------------
// Fragment helpers (CDNA5 WMMA 16x16x32 f16, wave32)
// ---------------------------------------------------------------------------

// A fragment (16x32 f16): lane&15 = M row; lane>=16 -> khalf=8.
// Elements 0..7  = K[kbase+khalf .. +7]   (contiguous)
// Elements 8..15 = K[kbase+16+khalf .. +7] (contiguous)
__device__ __forceinline__ v16h load_a_frag(const _Float16* row, int kbase, int lane) {
    int khalf = (lane >> 4) << 3;
    v8h lo = *reinterpret_cast<const v8h*>(row + kbase + khalf);
    v8h hi = *reinterpret_cast<const v8h*>(row + kbase + 16 + khalf);
    return __builtin_shufflevector(lo, hi, 0,1,2,3,4,5,6,7,8,9,10,11,12,13,14,15);
}

// B fragments pre-packed fragment-major: frag (kt*16+nt), lane -> 16 f16 (32B)
__device__ __forceinline__ v16h load_b_frag(const _Float16* pack, int kt, int nt, int lane) {
    return reinterpret_cast<const v16h*>(pack)[(kt * 16 + nt) * 32 + lane];
}

__device__ __forceinline__ v8f wmma_f16(v16h a, v16h b, v8f c) {
    return __builtin_amdgcn_wmma_f32_16x16x32_f16(false, a, false, b, (short)0, c, false, false);
}

__device__ __forceinline__ float silu_f(float x) { return x / (1.f + __expf(-x)); }

// ---------------------------------------------------------------------------
// Weight packing: W (K x 256 f32, row-major) -> B-fragment-major f16
// one thread per output f16 element; KT = K/32 k-tiles, 16 n-tiles
// ---------------------------------------------------------------------------
__global__ void pack_b_kernel(const float* __restrict__ W, _Float16* __restrict__ out,
                              int KT, int ld) {
    int idx = blockIdx.x * blockDim.x + threadIdx.x;
    int total = KT * 16 * 512;
    if (idx >= total) return;
    int f    = idx >> 9;
    int rem  = idx & 511;
    int lane = rem >> 4;
    int e    = rem & 15;
    int kt = f >> 4;
    int nt = f & 15;
    int col   = nt * 16 + (lane & 15);
    int khalf = (lane >> 4) << 3;
    int v = e >> 1;
    int klocal = (v < 4) ? (khalf + 2 * v + (e & 1)) : (16 + khalf + 2 * (v - 4) + (e & 1));
    int k = kt * 32 + klocal;
    out[idx] = (_Float16)W[k * ld + col];
}

// ---------------------------------------------------------------------------
// rep @ W_rep + b_rep  -> repterm (BS,256)
// ---------------------------------------------------------------------------
__global__ void rep_kernel(const float* __restrict__ rep, const float* __restrict__ Wr,
                           const float* __restrict__ br, float* __restrict__ repterm) {
    int b = blockIdx.x, col = threadIdx.x;
    const float* rp = rep + b * REP_NF;
    float s = br[col];
    for (int q = 0; q < REP_NF; ++q) s += rp[q] * Wr[q * HD + col];
    repterm[b * HD + col] = s;
}

// ---------------------------------------------------------------------------
// h0 = [xh[...,3:9]*nm, t] @ W_embed + b_embed + repterm ; also x0, pad rows
// grid BS*NPAD, block 256
// ---------------------------------------------------------------------------
__global__ void embed_kernel(const float* __restrict__ xh, const float* __restrict__ nm,
                             const float* __restrict__ tt, const float* __restrict__ We,
                             const float* __restrict__ be, const float* __restrict__ repterm,
                             float* __restrict__ hf32, _Float16* __restrict__ hpad,
                             float* __restrict__ x0) {
    int bi = blockIdx.x;
    int b = bi / NPAD, i = bi % NPAD;
    int col = threadIdx.x;
    if (i >= NN) { hpad[(b * NPAD + i) * HD + col] = (_Float16)0.f; return; }
    float m = nm[b * NN + i];
    const float* xr = xh + (b * NN + i) * 9;
    float s = be[col] + repterm[b * HD + col];
    #pragma unroll
    for (int q = 0; q < 6; ++q) s += (xr[3 + q] * m) * We[q * HD + col];
    s += tt[b] * We[6 * HD + col];
    hf32[(b * NN + i) * HD + col] = s;
    hpad[(b * NPAD + i) * HD + col] = (_Float16)s;
    if (col < 3) x0[(b * NN + i) * 3 + col] = xr[col] * m;
}

// ---------------------------------------------------------------------------
// radial distances, padded to NPAD columns. grid BS*NN, block 64
// ---------------------------------------------------------------------------
__global__ void radial_kernel(const float* __restrict__ x, float* __restrict__ dout) {
    int bi = blockIdx.x;
    int b = bi / NN, i = bi % NN;
    int j = threadIdx.x;
    if (j >= NPAD) return;
    float v = 0.f;
    if (j < NN) {
        const float* xi = x + (b * NN + i) * 3;
        const float* xj = x + (b * NN + j) * 3;
        float dx = xi[0] - xj[0], dy = xi[1] - xj[1], dz = xi[2] - xj[2];
        v = dx * dx + dy * dy + dz * dz;
    }
    dout[(b * NN + i) * NPAD + j] = v;
}

// ---------------------------------------------------------------------------
// Edge MLP + aggregation. One block per (b,i); 8 waves; wave owns 2 n-tiles.
// m = silu(silu([hi|hj|d|d0]@We1+be1)@We2+be2)*emask ; agg = sum_j m / 100
// ---------------------------------------------------------------------------
__global__ __launch_bounds__(256) void edge_mlp_kernel(
    const _Float16* __restrict__ hpad, const float* __restrict__ dcur,
    const float* __restrict__ d0p, const float* __restrict__ emask,
    const _Float16* __restrict__ W1p, const float* __restrict__ W1x,
    const float* __restrict__ b1, const _Float16* __restrict__ W2p,
    const float* __restrict__ b2, _Float16* __restrict__ aggf16) {
    __shared__ __align__(16) _Float16 smh[16 * HD];
    __shared__ float sm_em[NPAD], sm_d[NPAD], sm_d0[NPAD];
    int bi = blockIdx.x;
    int b = bi / NN, i = bi % NN;
    int tid = threadIdx.x, lane = tid & 31, wave = tid >> 5;
    for (int q = tid; q < NPAD; q += 256) {
        sm_em[q] = (q < NN) ? emask[(b * NN + i) * NN + q] : 0.f;
        sm_d[q]  = dcur[(b * NN + i) * NPAD + q];
        sm_d0[q] = d0p[(b * NN + i) * NPAD + q];
    }
    __syncthreads();
    int nt0 = wave * 2;
    int col0 = nt0 * 16 + (lane & 15), col1 = col0 + 16;
    int mhalf = (lane >> 4) << 3;
    float w512_0 = W1x[col0], w513_0 = W1x[HD + col0];
    float w512_1 = W1x[col1], w513_1 = W1x[HD + col1];
    float bb10 = b1[col0], bb11 = b1[col1];
    float bb20 = b2[col0], bb21 = b2[col1];
    const _Float16* hi_row = hpad + (b * NPAD + i) * HD;
    float agg0 = 0.f, agg1 = 0.f;
    for (int mt = 0; mt < 3; ++mt) {
        const _Float16* hj_row = hpad + (b * NPAD + mt * 16 + (lane & 15)) * HD;
        v8f c0 = {}; v8f c1 = {};
        #pragma unroll
        for (int kt = 0; kt < 16; ++kt) {
            v16h a = (kt < 8) ? load_a_frag(hi_row, kt * 32, lane)
                              : load_a_frag(hj_row, kt * 32 - HD, lane);
            c0 = wmma_f16(a, load_b_frag(W1p, kt, nt0, lane), c0);
            c1 = wmma_f16(a, load_b_frag(W1p, kt, nt0 + 1, lane), c1);
        }
        #pragma unroll
        for (int r = 0; r < 8; ++r) {
            int jr = mt * 16 + mhalf + r;
            float dv = sm_d[jr], dv0 = sm_d0[jr];
            smh[(mhalf + r) * HD + col0] = (_Float16)silu_f(c0[r] + bb10 + dv * w512_0 + dv0 * w513_0);
            smh[(mhalf + r) * HD + col1] = (_Float16)silu_f(c1[r] + bb11 + dv * w512_1 + dv0 * w513_1);
        }
        __syncthreads();
        const _Float16* arow = smh + (lane & 15) * HD;
        v8f e0 = {}; v8f e1 = {};
        #pragma unroll
        for (int kt = 0; kt < 8; ++kt) {
            v16h a = load_a_frag(arow, kt * 32, lane);
            e0 = wmma_f16(a, load_b_frag(W2p, kt, nt0, lane), e0);
            e1 = wmma_f16(a, load_b_frag(W2p, kt, nt0 + 1, lane), e1);
        }
        float s0 = 0.f, s1 = 0.f;
        #pragma unroll
        for (int r = 0; r < 8; ++r) {
            float em = sm_em[mt * 16 + mhalf + r];
            s0 += silu_f(e0[r] + bb20) * em;
            s1 += silu_f(e1[r] + bb21) * em;
        }
        s0 += __shfl_xor(s0, 16);
        s1 += __shfl_xor(s1, 16);
        agg0 += s0; agg1 += s1;
        __syncthreads();
    }
    if (lane < 16) {
        int node = b * NN + i;
        aggf16[node * HD + col0] = (_Float16)(agg0 * INV_NORM);
        aggf16[node * HD + col1] = (_Float16)(agg1 * INV_NORM);
    }
}

// ---------------------------------------------------------------------------
// Node MLP: h = (h + silu([h|agg]@Wn1+bn1)@Wn2+bn2) * nm
// grid (BS*NN)/16 blocks; each block = 16 node rows x 256 cols
// ---------------------------------------------------------------------------
__global__ __launch_bounds__(256) void node_mlp_kernel(
    const _Float16* __restrict__ hpad, const _Float16* __restrict__ aggf16,
    const float* __restrict__ nm,
    const _Float16* __restrict__ W1p, const float* __restrict__ b1,
    const _Float16* __restrict__ W2p, const float* __restrict__ b2,
    float* __restrict__ hf32, _Float16* __restrict__ hpad_out) {
    __shared__ __align__(16) _Float16 smh[16 * HD];
    __shared__ float sm_nm[16];
    int rowbase = blockIdx.x * 16;
    int tid = threadIdx.x, lane = tid & 31, wave = tid >> 5;
    if (tid < 16) sm_nm[tid] = nm[rowbase + tid];
    __syncthreads();
    int nt0 = wave * 2;
    int col0 = nt0 * 16 + (lane & 15), col1 = col0 + 16;
    int mhalf = (lane >> 4) << 3;
    int mrow = rowbase + (lane & 15);
    int bb = mrow / NN, ii = mrow % NN;
    const _Float16* ha = hpad + (bb * NPAD + ii) * HD;
    const _Float16* ga = aggf16 + mrow * HD;
    float bb10 = b1[col0], bb11 = b1[col1], bb20 = b2[col0], bb21 = b2[col1];
    v8f c0 = {}; v8f c1 = {};
    #pragma unroll
    for (int kt = 0; kt < 16; ++kt) {
        v16h a = (kt < 8) ? load_a_frag(ha, kt * 32, lane) : load_a_frag(ga, kt * 32 - HD, lane);
        c0 = wmma_f16(a, load_b_frag(W1p, kt, nt0, lane), c0);
        c1 = wmma_f16(a, load_b_frag(W1p, kt, nt0 + 1, lane), c1);
    }
    #pragma unroll
    for (int r = 0; r < 8; ++r) {
        smh[(mhalf + r) * HD + col0] = (_Float16)silu_f(c0[r] + bb10);
        smh[(mhalf + r) * HD + col1] = (_Float16)silu_f(c1[r] + bb11);
    }
    __syncthreads();
    const _Float16* arow = smh + (lane & 15) * HD;
    v8f e0 = {}; v8f e1 = {};
    #pragma unroll
    for (int kt = 0; kt < 8; ++kt) {
        v16h a = load_a_frag(arow, kt * 32, lane);
        e0 = wmma_f16(a, load_b_frag(W2p, kt, nt0, lane), e0);
        e1 = wmma_f16(a, load_b_frag(W2p, kt, nt0 + 1, lane), e1);
    }
    #pragma unroll
    for (int r = 0; r < 8; ++r) {
        int row = rowbase + mhalf + r;
        float m = sm_nm[mhalf + r];
        int b2i = row / NN, i2 = row % NN;
        float h0 = (hf32[row * HD + col0] + e0[r] + bb20) * m;
        float h1 = (hf32[row * HD + col1] + e1[r] + bb21) * m;
        hf32[row * HD + col0] = h0;
        hf32[row * HD + col1] = h1;
        hpad_out[(b2i * NPAD + i2) * HD + col0] = (_Float16)h0;
        hpad_out[(b2i * NPAD + i2) * HD + col1] = (_Float16)h1;
    }
}

// ---------------------------------------------------------------------------
// Coordinate MLP: phi = silu(silu(e_in@Wc1+bc1)@Wc2+bc2)@Wc3
// x_out = (x_in + sum_j cdiff*phi*emask / 100) * nm ; one block per (b,i)
// ---------------------------------------------------------------------------
__global__ __launch_bounds__(256) void coord_mlp_kernel(
    const _Float16* __restrict__ hpad, const float* __restrict__ dcur,
    const float* __restrict__ d0p, const float* __restrict__ emask,
    const float* __restrict__ nm,
    const _Float16* __restrict__ W1p, const float* __restrict__ W1x,
    const float* __restrict__ b1, const _Float16* __restrict__ W2p,
    const float* __restrict__ b2, const float* __restrict__ wc3,
    const float* __restrict__ xin, float* __restrict__ xout) {
    __shared__ __align__(16) _Float16 smh[16 * HD];
    __shared__ float sm_em[NPAD], sm_d[NPAD], sm_d0[NPAD], sm_phi[NPAD];
    __shared__ float sm_tr[NPAD * 3];
    int bi = blockIdx.x;
    int b = bi / NN, i = bi % NN;
    int tid = threadIdx.x, lane = tid & 31, wave = tid >> 5;
    for (int q = tid; q < NPAD; q += 256) {
        sm_em[q] = (q < NN) ? emask[(b * NN + i) * NN + q] : 0.f;
        sm_d[q]  = dcur[(b * NN + i) * NPAD + q];
        sm_d0[q] = d0p[(b * NN + i) * NPAD + q];
        sm_phi[q] = 0.f;
    }
    __syncthreads();
    int nt0 = wave * 2;
    int col0 = nt0 * 16 + (lane & 15), col1 = col0 + 16;
    int mhalf = (lane >> 4) << 3;
    float w512_0 = W1x[col0], w513_0 = W1x[HD + col0];
    float w512_1 = W1x[col1], w513_1 = W1x[HD + col1];
    float bb10 = b1[col0], bb11 = b1[col1];
    float bb20 = b2[col0], bb21 = b2[col1];
    float wc30 = wc3[col0], wc31 = wc3[col1];
    const _Float16* hi_row = hpad + (b * NPAD + i) * HD;
    for (int mt = 0; mt < 3; ++mt) {
        const _Float16* hj_row = hpad + (b * NPAD + mt * 16 + (lane & 15)) * HD;
        v8f c0 = {}; v8f c1 = {};
        #pragma unroll
        for (int kt = 0; kt < 16; ++kt) {
            v16h a = (kt < 8) ? load_a_frag(hi_row, kt * 32, lane)
                              : load_a_frag(hj_row, kt * 32 - HD, lane);
            c0 = wmma_f16(a, load_b_frag(W1p, kt, nt0, lane), c0);
            c1 = wmma_f16(a, load_b_frag(W1p, kt, nt0 + 1, lane), c1);
        }
        #pragma unroll
        for (int r = 0; r < 8; ++r) {
            int jr = mt * 16 + mhalf + r;
            float dv = sm_d[jr], dv0 = sm_d0[jr];
            smh[(mhalf + r) * HD + col0] = (_Float16)silu_f(c0[r] + bb10 + dv * w512_0 + dv0 * w513_0);
            smh[(mhalf + r) * HD + col1] = (_Float16)silu_f(c1[r] + bb11 + dv * w512_1 + dv0 * w513_1);
        }
        __syncthreads();
        const _Float16* arow = smh + (lane & 15) * HD;
        v8f e0 = {}; v8f e1 = {};
        #pragma unroll
        for (int kt = 0; kt < 8; ++kt) {
            v16h a = load_a_frag(arow, kt * 32, lane);
            e0 = wmma_f16(a, load_b_frag(W2p, kt, nt0, lane), e0);
            e1 = wmma_f16(a, load_b_frag(W2p, kt, nt0 + 1, lane), e1);
        }
        float p[8];
        #pragma unroll
        for (int r = 0; r < 8; ++r)
            p[r] = silu_f(e0[r] + bb20) * wc30 + silu_f(e1[r] + bb21) * wc31;
        #pragma unroll
        for (int mk = 1; mk <= 8; mk <<= 1) {
            #pragma unroll
            for (int r = 0; r < 8; ++r) p[r] += __shfl_xor(p[r], mk);
        }
        if ((lane & 15) == 0) {
            #pragma unroll
            for (int r = 0; r < 8; ++r) atomicAdd(&sm_phi[mt * 16 + mhalf + r], p[r]);
        }
        __syncthreads();
    }
    if (tid < NPAD) {
        int j = tid;
        float phi = sm_phi[j], em = sm_em[j];
        float inv = rsqrtf(sm_d[j] + 1e-8f);
        int jj = (j < NN) ? j : 0;
        const float* xi = xin + (b * NN + i) * 3;
        const float* xj = xin + (b * NN + jj) * 3;
        #pragma unroll
        for (int c = 0; c < 3; ++c) sm_tr[j * 3 + c] = (xi[c] - xj[c]) * inv * phi * em;
    }
    __syncthreads();
    if (tid < 3) {
        float s = 0.f;
        for (int j = 0; j < NPAD; ++j) s += sm_tr[j * 3 + tid];
        float m = nm[b * NN + i];
        xout[(b * NN + i) * 3 + tid] = (xin[(b * NN + i) * 3 + tid] + s * INV_NORM) * m;
    }
}

// ---------------------------------------------------------------------------
// Per-batch velocity sums + node counts
// ---------------------------------------------------------------------------
__global__ void stats_kernel(const float* __restrict__ xf, const float* __restrict__ x0,
                             const float* __restrict__ nm, float* __restrict__ stats) {
    __shared__ float sv[64][4];
    int b = blockIdx.x, t = threadIdx.x;
    float c0 = 0.f, c1 = 0.f, c2 = 0.f, cn = 0.f;
    if (t < NN) {
        float m = nm[b * NN + t];
        const float* a = xf + (b * NN + t) * 3;
        const float* z = x0 + (b * NN + t) * 3;
        c0 = (a[0] - z[0]) * m; c1 = (a[1] - z[1]) * m; c2 = (a[2] - z[2]) * m; cn = m;
    }
    sv[t][0] = c0; sv[t][1] = c1; sv[t][2] = c2; sv[t][3] = cn;
    __syncthreads();
    if (t == 0) {
        float s0 = 0.f, s1 = 0.f, s2 = 0.f, s3 = 0.f;
        for (int q = 0; q < 64; ++q) { s0 += sv[q][0]; s1 += sv[q][1]; s2 += sv[q][2]; s3 += sv[q][3]; }
        stats[b * 4 + 0] = s0; stats[b * 4 + 1] = s1; stats[b * 4 + 2] = s2; stats[b * 4 + 3] = s3;
    }
}

// ---------------------------------------------------------------------------
// Output: [vel(3) | (h@W_out+b_out)[:6]] per node ; grid BS*NN, block 64
// ---------------------------------------------------------------------------
__global__ void out_kernel(const float* __restrict__ hf32, const float* __restrict__ Wo,
                           const float* __restrict__ bo, const float* __restrict__ xf,
                           const float* __restrict__ x0, const float* __restrict__ nm,
                           const float* __restrict__ stats, float* __restrict__ out) {
    int bi = blockIdx.x;
    int b = bi / NN;
    int t = threadIdx.x;
    float m = nm[bi];
    if (t < 6) {
        const float* hr = hf32 + bi * HD;
        float s = bo[t];
        for (int k = 0; k < HD; ++k) s += hr[k] * Wo[k * 7 + t];
        out[bi * 9 + 3 + t] = s * m;
    } else if (t < 9) {
        int c = t - 6;
        float Nnv = stats[b * 4 + 3];
        float mean = stats[b * 4 + c] / Nnv;
        float vel = (xf[bi * 3 + c] - x0[bi * 3 + c]) * m;
        out[bi * 9 + c] = (vel - mean) * m;
    }
}

// ---------------------------------------------------------------------------
extern "C" void kernel_launch(void* const* d_in, const int* in_sizes, int n_in,
                              void* d_out, int out_size, void* d_ws, size_t ws_size,
                              hipStream_t stream) {
    (void)in_sizes; (void)n_in; (void)out_size; (void)ws_size;
    const float* t_in      = (const float*)d_in[0];
    const float* xh        = (const float*)d_in[1];
    const float* node_mask = (const float*)d_in[2];
    const float* edge_mask = (const float*)d_in[3];
    const float* rep       = (const float*)d_in[4];
    const float* W_embed   = (const float*)d_in[5];
    const float* b_embed   = (const float*)d_in[6];
    const float* W_rep     = (const float*)d_in[7];
    const float* b_rep     = (const float*)d_in[8];
    const float* We1       = (const float*)d_in[9];
    const float* be1       = (const float*)d_in[10];
    const float* We2       = (const float*)d_in[11];
    const float* be2       = (const float*)d_in[12];
    const float* Wn1       = (const float*)d_in[13];
    const float* bn1       = (const float*)d_in[14];
    const float* Wn2       = (const float*)d_in[15];
    const float* bn2       = (const float*)d_in[16];
    const float* Wc1       = (const float*)d_in[17];
    const float* bc1       = (const float*)d_in[18];
    const float* Wc2       = (const float*)d_in[19];
    const float* bc2       = (const float*)d_in[20];
    const float* Wc3       = (const float*)d_in[21];
    const float* W_out     = (const float*)d_in[22];
    const float* b_out     = (const float*)d_in[23];

    char* ws = (char*)d_ws;
    auto carve = [&](size_t bytes) -> char* {
        char* p = ws;
        ws += (bytes + 255) & ~(size_t)255;
        return p;
    };

    _Float16* hpad    = (_Float16*)carve((size_t)BSZ * NPAD * HD * 2);
    float*    hf32    = (float*)   carve((size_t)BSZ * NN * HD * 4);
    _Float16* aggf16  = (_Float16*)carve((size_t)BSZ * NN * HD * 2);
    float*    d0p     = (float*)   carve((size_t)BSZ * NN * NPAD * 4);
    float*    dcurp   = (float*)   carve((size_t)BSZ * NN * NPAD * 4);
    float*    repterm = (float*)   carve((size_t)BSZ * HD * 4);
    float*    x0b     = (float*)   carve((size_t)BSZ * NN * 3 * 4);
    float*    xA      = (float*)   carve((size_t)BSZ * NN * 3 * 4);
    float*    xB      = (float*)   carve((size_t)BSZ * NN * 3 * 4);
    float*    statb   = (float*)   carve((size_t)BSZ * 4 * 4);

    const int PK16 = 16 * 16 * 512;   // elements for K=512 pack
    const int PK8  = 8 * 16 * 512;    // elements for K=256 pack
    _Float16* We1p = (_Float16*)carve((size_t)GG * PK16 * 2);
    _Float16* We2p = (_Float16*)carve((size_t)GG * PK8 * 2);
    _Float16* Wn1p = (_Float16*)carve((size_t)GG * PK16 * 2);
    _Float16* Wn2p = (_Float16*)carve((size_t)GG * PK8 * 2);
    _Float16* Wc1p = (_Float16*)carve((size_t)NLAYERS * PK16 * 2);
    _Float16* Wc2p = (_Float16*)carve((size_t)NLAYERS * PK8 * 2);

    // pack weights to WMMA B-fragment layout (f16)
    for (int k = 0; k < GG; ++k) {
        pack_b_kernel<<<(PK16 + 255) / 256, 256, 0, stream>>>(We1 + (size_t)k * 514 * HD, We1p + (size_t)k * PK16, 16, HD);
        pack_b_kernel<<<(PK8 + 255) / 256, 256, 0, stream>>>(We2 + (size_t)k * HD * HD, We2p + (size_t)k * PK8, 8, HD);
        pack_b_kernel<<<(PK16 + 255) / 256, 256, 0, stream>>>(Wn1 + (size_t)k * 512 * HD, Wn1p + (size_t)k * PK16, 16, HD);
        pack_b_kernel<<<(PK8 + 255) / 256, 256, 0, stream>>>(Wn2 + (size_t)k * HD * HD, Wn2p + (size_t)k * PK8, 8, HD);
    }
    for (int l = 0; l < NLAYERS; ++l) {
        pack_b_kernel<<<(PK16 + 255) / 256, 256, 0, stream>>>(Wc1 + (size_t)l * 514 * HD, Wc1p + (size_t)l * PK16, 16, HD);
        pack_b_kernel<<<(PK8 + 255) / 256, 256, 0, stream>>>(Wc2 + (size_t)l * HD * HD, Wc2p + (size_t)l * PK8, 8, HD);
    }

    rep_kernel<<<BSZ, 256, 0, stream>>>(rep, W_rep, b_rep, repterm);
    embed_kernel<<<BSZ * NPAD, 256, 0, stream>>>(xh, node_mask, t_in, W_embed, b_embed,
                                                 repterm, hf32, hpad, x0b);
    radial_kernel<<<BSZ * NN, 64, 0, stream>>>(x0b, d0p);

    const float* xin = x0b;
    float* xbufs[2] = {xA, xB};
    for (int l = 0; l < NLAYERS; ++l) {
        const float* dptr = d0p;
        if (l > 0) {
            radial_kernel<<<BSZ * NN, 64, 0, stream>>>(xin, dcurp);
            dptr = dcurp;
        }
        for (int s = 0; s < INV_SUB; ++s) {
            int k = l * INV_SUB + s;
            edge_mlp_kernel<<<BSZ * NN, 256, 0, stream>>>(
                hpad, dptr, d0p, edge_mask,
                We1p + (size_t)k * PK16, We1 + (size_t)k * 514 * HD + 512 * HD, be1 + (size_t)k * HD,
                We2p + (size_t)k * PK8, be2 + (size_t)k * HD, aggf16);
            node_mlp_kernel<<<(BSZ * NN) / 16, 256, 0, stream>>>(
                hpad, aggf16, node_mask,
                Wn1p + (size_t)k * PK16, bn1 + (size_t)k * HD,
                Wn2p + (size_t)k * PK8, bn2 + (size_t)k * HD, hf32, hpad);
        }
        float* xout = xbufs[l & 1];
        coord_mlp_kernel<<<BSZ * NN, 256, 0, stream>>>(
            hpad, dptr, d0p, edge_mask, node_mask,
            Wc1p + (size_t)l * PK16, Wc1 + (size_t)l * 514 * HD + 512 * HD, bc1 + (size_t)l * HD,
            Wc2p + (size_t)l * PK8, bc2 + (size_t)l * HD, Wc3 + (size_t)l * HD,
            xin, xout);
        xin = xout;
    }

    stats_kernel<<<BSZ, 64, 0, stream>>>(xin, x0b, node_mask, statb);
    out_kernel<<<BSZ * NN, 64, 0, stream>>>(hf32, W_out, b_out, xin, x0b, node_mask,
                                            statb, (float*)d_out);
}
